// MarkovRandomField_3934190044188
// MI455X (gfx1250) — matrix operand order repaired and checked by reading, not compiled, not run on hardware
//
#include <hip/hip_runtime.h>

#define NN 4096
#define LL 343
#define LP 352          // padded row stride for f16 A
#define INFV 1e30f
#define UPT 704         // up_kernel threads: 2 reduction groups x 352
#define ABYTES ((size_t)LL * LP * 2)   // 241472 bytes, 16B-divisible
#define ACHUNKS (ABYTES / 16)          // 15092 b128 chunks

typedef __attribute__((ext_vector_type(16))) _Float16 v16h;
typedef __attribute__((ext_vector_type(8)))  float    v8f;

// ---------------- Prim's MST: 1 persistent workgroup, all state in LDS ----------------
__global__ __launch_bounds__(1024) void prim_kernel(const float* __restrict__ dist,
                                                    int* __restrict__ parent,
                                                    int* __restrict__ order,
                                                    float* __restrict__ edgeW) {
    __shared__ float best[NN];
    __shared__ int   par[NN];
    __shared__ unsigned char inT[NN];
    __shared__ float sval[1024];
    __shared__ int   sidx[1024];
    const int t = threadIdx.x;
    for (int j = t; j < NN; j += 1024) {
        best[j] = dist[j];          // row 0 (root = 0)
        par[j]  = 0;
        inT[j]  = (j == 0) ? 1 : 0;
    }
    if (t == 0) order[0] = 0;
    __syncthreads();
    for (int i = 1; i < NN; ++i) {
        float bv = INFV; int bi = 0;
        for (int j = t; j < NN; j += 1024) {
            float v = inT[j] ? INFV : best[j];
            if (v < bv) { bv = v; bi = j; }
        }
        sval[t] = bv; sidx[t] = bi;
        __syncthreads();
        for (int s = 512; s > 0; s >>= 1) {
            if (t < s) {
                float ov = sval[t + s]; int oi = sidx[t + s];
                if (ov < sval[t] || (ov == sval[t] && oi < sidx[t])) { sval[t] = ov; sidx[t] = oi; }
            }
            __syncthreads();
        }
        const int nxt = sidx[0];
        if (t == 0) { inT[nxt] = 1; order[i] = nxt; }
        __syncthreads();
        const float* drow = dist + (size_t)nxt * NN;
        for (int j = t; j < NN; j += 1024) {
            float dn = drow[j];
            if (!inT[j] && dn < best[j]) { best[j] = dn; par[j] = nxt; }
        }
        __syncthreads();
    }
    for (int j = t; j < NN; j += 1024) {
        parent[j] = par[j];
        edgeW[j]  = expf(-dist[(size_t)j * NN + par[j]] / (2.0f * 0.3f * 0.3f));
    }
}

// ---------------- A (f32 343x343) -> f16 padded 343x352 ----------------
__global__ void a2h_kernel(const float* __restrict__ A, _Float16* __restrict__ A16) {
    size_t i = (size_t)blockIdx.x * 256 + threadIdx.x;
    if (i < (size_t)LL * LP) {
        int l = (int)(i / LP), lp = (int)(i % LP);
        A16[i] = (_Float16)(lp < LL ? A[(size_t)l * LL + lp] : 0.0f);
    }
}

// ---------------- belief init: AU * unary (AU = 1.0) ----------------
__global__ void initb_kernel(const float* __restrict__ u, float* __restrict__ b) {
    size_t i = (size_t)blockIdx.x * 256 + threadIdx.x;
    if (i < (size_t)NN * LL) b[i] = 1.0f * u[i];
}

// ---------------- upward Viterbi scan: LDS-resident A, async staging, 2-way split reduce ----
// Dynamic LDS layout: sA[LL*LP] f16 (241472B) | sB[LP] f32 | sM[UPT] f32 | sArg[UPT] i32
__global__ __launch_bounds__(UPT) void up_kernel(const _Float16* __restrict__ A16,
                                                 const int* __restrict__ parent,
                                                 const int* __restrict__ order,
                                                 const float* __restrict__ edgeW,
                                                 float* __restrict__ belief,
                                                 unsigned short* __restrict__ ptrbuf) {
    extern __shared__ char smem[];
    _Float16* sA  = (_Float16*)smem;
    float*    sB  = (float*)(smem + ABYTES);
    float*    sM  = (float*)(smem + ABYTES + LP * 4);
    int*      sArg = (int*)(smem + ABYTES + LP * 4 + UPT * 4);
    const int t = threadIdx.x;

    // one-time bulk async stage of the whole 241KB f16 transition matrix into WGP LDS
    for (size_t idx = t; idx < ACHUNKS; idx += UPT) {
        unsigned long long ga = (unsigned long long)(const void*)((const char*)A16 + idx * 16);
        unsigned lo = (unsigned)(size_t)(smem + idx * 16);
        asm volatile("global_load_async_to_lds_b128 %0, %1, off" :: "v"(lo), "v"(ga) : "memory");
    }
    asm volatile("s_wait_asynccnt 0" ::: "memory");
    __syncthreads();

    const int col = t % 352;
    const int grp = t / 352;                 // 2-way split of the l-reduction
    const int l0  = grp ? 172 : 0;
    const int l1  = grp ? LL  : 172;

    for (int step = 0; step < NN; ++step) {
        const int   c = order[NN - 1 - step];
        const int   p = parent[c];
        const float w = edgeW[c];
        // stage belief[c] row (1.4KB) into LDS via the async path
        if (t < LL) {
            unsigned long long ga = (unsigned long long)(const void*)(belief + (size_t)c * LL + t);
            unsigned lo = (unsigned)(size_t)(&sB[t]);
            asm volatile("global_load_async_to_lds_b32 %0, %1, off" :: "v"(lo), "v"(ga) : "memory");
        }
        asm volatile("s_wait_asynccnt 0" ::: "memory");
        __syncthreads();

        float m = -INFV; int am = 0;
        if (col < LL) {
            for (int l = l0; l < l1; ++l) {
                float s = fmaf(w, (float)sA[(size_t)l * LP + col], sB[l]);
                if (s > m) { m = s; am = l; }        // first-argmax within group
            }
        }
        sM[t] = m; sArg[t] = am;
        __syncthreads();

        if (t < LL) {
            float m1 = sM[t + 352]; int a1 = sArg[t + 352];
            if (m1 > m) { m = m1; am = a1; }         // group0 owns smaller l => ties keep group0
            ptrbuf[(size_t)c * LL + t] = (unsigned short)am;
            if (c != 0) belief[(size_t)p * LL + t] += m;   // root sends nothing
        }
        asm volatile("s_wait_storecnt 0" ::: "memory");
        __syncthreads();
    }
}

// ---------------- root argmax + sequential backtracking ----------------
__global__ __launch_bounds__(512) void down_kernel(const float* __restrict__ belief,
                                                   const unsigned short* __restrict__ ptrbuf,
                                                   const int* __restrict__ parent,
                                                   const int* __restrict__ order,
                                                   int* __restrict__ labels) {
    __shared__ float sval[512];
    __shared__ int   sidx[512];
    const int t = threadIdx.x;
    float bv = -INFV; int bi = 0;
    for (int l = t; l < LL; l += 512) { float v = belief[l]; if (v > bv) { bv = v; bi = l; } }
    sval[t] = bv; sidx[t] = bi;
    __syncthreads();
    for (int s = 256; s > 0; s >>= 1) {
        if (t < s) {
            if (sval[t + s] > sval[t] || (sval[t + s] == sval[t] && sidx[t + s] < sidx[t])) {
                sval[t] = sval[t + s]; sidx[t] = sidx[t + s];
            }
        }
        __syncthreads();
    }
    if (t == 0) {
        labels[0] = sidx[0];                          // root == order[0] == 0
        for (int i = 1; i < NN; ++i) {
            int c = order[i];
            labels[c] = (int)ptrbuf[(size_t)c * LL + labels[parent[c]]];
        }
    }
}

// ---------------- one-hot epilogue via the matrix pipe: onehot = S x I (WMMA f16) ----------------
__global__ __launch_bounds__(256) void out_kernel(const int* __restrict__ labels,
                                                  float* __restrict__ out) {
    const int wave = threadIdx.x >> 5;
    const int lane = threadIdx.x & 31;
    const int tile = blockIdx.x * 8 + wave;     // 5632 tiles = 256 node-tiles x 22 col-tiles
    const int CT = 22;
    const int tn = tile / CT;
    const int tc = tile % CT;
    const int row = lane & 15;
    const int lab = labels[tn * 16 + row];
    const int K   = lab - tc * 16;

    v16h a, b;
    for (int e = 0; e < 16; ++e) { a[e] = (_Float16)0; b[e] = (_Float16)0; }
    if (K >= 0 && K < 16) {
        if (lane < 16) a[K] = (_Float16)1;
    }
    if (lane < 16) b[lane] = (_Float16)1;      // identity, K<16 live

    v8f c = {};
    v8f d = __builtin_amdgcn_wmma_f32_16x16x32_f16(false, a, false, b, (short)0, c, false, false);

    const int col  = tc * 16 + (lane & 15);
    const int madd = (lane >> 4) * 8;
    if (col < LL) {
        for (int r = 0; r < 8; ++r) {
            int m = r + madd;
            out[(size_t)(tn * 16 + m) * LL + col] = 2.0e5f * d[r] - 1.0e5f;
        }
    }
}

extern "C" void kernel_launch(void* const* d_in, const int* in_sizes, int n_in,
                              void* d_out, int out_size, void* d_ws, size_t ws_size,
                              hipStream_t stream) {
    const float* unary = (const float*)d_in[0];   // (N, L) f32
    const float* dist  = (const float*)d_in[1];   // (N, N) f32
    const float* A     = (const float*)d_in[2];   // (L, L) f32
    float* out = (float*)d_out;                   // (N, L) f32

    char* w = (char*)d_ws;
    auto take = [&](size_t bytes) { char* p = w; w += (bytes + 255) & ~(size_t)255; return p; };
    float*          belief = (float*)          take((size_t)NN * LL * sizeof(float));
    unsigned short* ptrbuf = (unsigned short*) take((size_t)NN * LL * sizeof(unsigned short));
    _Float16*       A16    = (_Float16*)       take(ABYTES);
    int*            parent = (int*)            take((size_t)NN * sizeof(int));
    int*            order  = (int*)            take((size_t)NN * sizeof(int));
    int*            labels = (int*)            take((size_t)NN * sizeof(int));
    float*          edgeW  = (float*)          take((size_t)NN * sizeof(float));

    const size_t upLds = ABYTES + (size_t)LP * 4 + (size_t)UPT * 4 * 2;  // ~248.5KB < 320KB WGP LDS

    prim_kernel <<<1, 1024, 0, stream>>>(dist, parent, order, edgeW);
    a2h_kernel  <<<((size_t)LL * LP + 255) / 256, 256, 0, stream>>>(A, A16);
    initb_kernel<<<((size_t)NN * LL + 255) / 256, 256, 0, stream>>>(unary, belief);
    up_kernel   <<<1, UPT, upLds, stream>>>(A16, parent, order, edgeW, belief, ptrbuf);
    down_kernel <<<1, 512, 0, stream>>>(belief, ptrbuf, parent, order, labels);
    out_kernel  <<<704, 256, 0, stream>>>(labels, out);   // 704*8 == 256*22 tiles exactly
}
